// Encoder_6614249635907
// MI455X (gfx1250) — compile-verified, matrix operand order checked by reading, not run on hardware
//
#include <hip/hip_runtime.h>

// ---------------- types ----------------
typedef __bf16 bf16_t;
typedef __attribute__((ext_vector_type(16))) __bf16 v16bf;
typedef __attribute__((ext_vector_type(8)))  __bf16 v8bf;
typedef __attribute__((ext_vector_type(8)))  float  v8f;
typedef __attribute__((ext_vector_type(4)))  unsigned int u32x4;
typedef __attribute__((ext_vector_type(8)))  int i32x8;
typedef __attribute__((ext_vector_type(4)))  int i32x4;

#define BB 1024
#define TT 128
#define NN 256
#define MM 128      // hidden
#define GM 512      // 4*M
#define HS_OFF 256  // hs lives in scratch cols [256..383]

#if __has_builtin(__builtin_amdgcn_tensor_load_to_lds)
#define HAVE_TDM 1
#else
#define HAVE_TDM 0
#endif

// ---------------- fast device math (gfx1250 native trans ops) ----------------
__device__ __forceinline__ float fexp2(float x) { return __builtin_amdgcn_exp2f(x); }
__device__ __forceinline__ float frcp(float x)  { return __builtin_amdgcn_rcpf(x); }
__device__ __forceinline__ float fsig(float x)  { return frcp(1.0f + fexp2(-1.44269504f * x)); }
__device__ __forceinline__ float ftanh(float x) {
#if __has_builtin(__builtin_amdgcn_tanhf)
  return __builtin_amdgcn_tanhf(x);           // v_tanh_f32
#else
  float e = fexp2(2.88539008f * x);           // e^(2x)
  return 1.0f - 2.0f * frcp(e + 1.0f);
#endif
}

__device__ __forceinline__ void wait_tensorcnt0() {
#if __has_builtin(__builtin_amdgcn_s_wait_tensorcnt)
  __builtin_amdgcn_s_wait_tensorcnt(0);
#else
  asm volatile("s_wait_tensorcnt 0x0" ::: "memory");
#endif
}

// ---------------- WMMA fragment loader ----------------
// 16-bit A/B fragment, 16x32 tile, row-major source with leading dim `ld` (halves).
__device__ __forceinline__ v16bf load_frag(const bf16_t* base, int ld, int kb, int lane) {
  const bf16_t* p = base + (size_t)(lane & 15) * ld + kb + ((lane >> 4) << 3);
  union { v16bf v; v8bf h[2]; } u;
  u.h[0] = *(const v8bf*)(p);
  u.h[1] = *(const v8bf*)(p + 16);
  return u.v;
}

__device__ __forceinline__ v8f wmma_bf16(v16bf a, v16bf b, v8f c) {
  return __builtin_amdgcn_wmma_f32_16x16x32_bf16(false, a, false, b, (short)0, c, false, false);
}

// ---------------- kernel 1: weights -> bf16, transposed [col][k] ----------------
__global__ __launch_bounds__(256) void prep_weights(
    const float* __restrict__ We, const float* __restrict__ Ue,
    const float* __restrict__ Wx, const float* __restrict__ Wh,
    bf16_t* __restrict__ UeT, bf16_t* __restrict__ WeT,
    bf16_t* __restrict__ WxT, bf16_t* __restrict__ WhT) {
  int i0 = blockIdx.x * blockDim.x + threadIdx.x;
  int stride = gridDim.x * blockDim.x;
  for (int i = i0; i < 128 * 128; i += stride) { // UeT[s][t] = Ue[t][s]
    int s = i >> 7, t = i & 127;
    UeT[i] = (bf16_t)Ue[t * 128 + s];
  }
  for (int i = i0; i < 128 * 256; i += stride) { // WeT[s][k] = We[k][s], We:(256,128)
    int s = i >> 8, k = i & 255;
    WeT[i] = (bf16_t)We[k * 128 + s];
  }
  for (int i = i0; i < 512 * 256; i += stride) { // WxT[j][k] = Wx[k][j], Wx:(256,512)
    int j = i >> 8, k = i & 255;
    WxT[i] = (bf16_t)Wx[k * 512 + j];
  }
  for (int i = i0; i < 512 * 128; i += stride) { // WhT[j][k] = Wh[k][j], Wh:(128,512)
    int j = i >> 7, k = i & 127;
    WhT[i] = (bf16_t)Wh[k * 512 + j];
  }
}

// ---------------- kernel 2: ux[b,n,s] = sum_t X[b,t,n]*Ue[t,s] + bu[s]  (bf16 out) ---
__global__ __launch_bounds__(256) void ux_gemm(
    const float* __restrict__ X, const float* __restrict__ bu,
    const bf16_t* __restrict__ UeT, bf16_t* __restrict__ ux) {
  __shared__ __align__(16) bf16_t Xs[16][TT]; // A tile (16 rows x K=128), bf16

  int tid  = threadIdx.x;
  int lane = tid & 31;
  int wave = tid >> 5;
  int r0 = blockIdx.x * 16;         // global row = b*N + n
  int b  = r0 >> 8;                 // N = 256
  int n0 = r0 & 255;

  for (int idx = tid; idx < 16 * TT; idx += 256) {
    int i = idx & 15, t = idx >> 4;
    Xs[i][t] = (bf16_t)X[((size_t)b * TT + t) * NN + n0 + i];
  }
  __syncthreads();

  int s0 = wave * 16;
  v8f acc = {};
#pragma unroll
  for (int kc = 0; kc < 4; ++kc) {
    v16bf a  = load_frag(&Xs[0][0], TT, kc * 32, lane);
    v16bf bb = load_frag(UeT + (size_t)s0 * TT, TT, kc * 32, lane);
    acc = wmma_bf16(a, bb, acc);
  }
  int s = s0 + (lane & 15);
  float bus = bu[s];
#pragma unroll
  for (int v = 0; v < 8; ++v) {
    int m = v + ((lane >> 4) << 3);
    ux[(size_t)(r0 + m) * TT + s] = (bf16_t)(acc[v] + bus);
  }
}

// ---------------- kernel 3: persistent recurrent scan -----------------------------
// 64 blocks x 512 threads (16 waves); each block owns 16 batch rows for all T steps.
__global__ __launch_bounds__(512) void darnn_scan(
    const float* __restrict__ X,  const float* __restrict__ be,
    const float* __restrict__ ve, const float* __restrict__ bv,
    const float* __restrict__ gb, const bf16_t* __restrict__ ux,
    const bf16_t* __restrict__ WeT, const bf16_t* __restrict__ WxT,
    const bf16_t* __restrict__ WhT, float* __restrict__ out) {
  __shared__ __align__(16) float  c_s[16][MM];        // cell state, f32        (8KB)
  __shared__ __align__(16) bf16_t hc_b[16][2 * MM];   // [h|c] bf16 A-frags     (8KB)
  __shared__ __align__(16) bf16_t xt_b[16][NN];       // x_tilde bf16 A-frags   (8KB)
  __shared__ __align__(16) float  scratch[16][GM];    // e | hs | gates         (32KB)
  __shared__ __align__(16) float  Xs_f[16][NN];       // TDM-staged x_t tile    (16KB)
  __shared__ float ve_s[TT];

  int tid  = threadIdx.x;
  int lane = tid & 31;
  int wave = tid >> 5;
  int b0   = blockIdx.x * 16;

  // init state
  for (int i = tid; i < 16 * MM; i += 512) c_s[i >> 7][i & 127] = 0.0f;
  for (int i = tid; i < 16 * 2 * MM; i += 512) hc_b[i >> 8][i & 255] = (bf16_t)0.0f;
  if (tid < TT) ve_s[tid] = ve[tid];
  float bvv = bv[0];
  __syncthreads();

  for (int t = 0; t < TT; ++t) {
#if HAVE_TDM
    // ---- (0) wave 15: TDM-stage x_t tile (16 rows x 256 f32, row stride T*N)
    if (wave == 15) {
      unsigned lds_off = (unsigned)(uintptr_t)(&Xs_f[0][0]);
      unsigned long long ga =
          (unsigned long long)(uintptr_t)X +
          4ull * ((size_t)b0 * TT * NN + (size_t)t * NN);
      u32x4 g0;
      g0[0] = 1u;                                          // count=1 (valid user D#)
      g0[1] = lds_off;                                     // lds_addr (bytes)
      g0[2] = (unsigned)(ga & 0xFFFFFFFFu);                // global_addr[31:0]
      g0[3] = (unsigned)((ga >> 32) & 0x1FFFFFFu) | (2u << 30); // addr[56:32] | type=2
      i32x8 g1 = {};
      g1[0] = (2 << 16);                  // data_size = 4B
      g1[1] = (int)(256u << 16);          // tensor_dim0 = 256 (bits 79:48, low half)
      g1[2] = (int)(16u << 16);           // tensor_dim1 = 16  (bits 111:80, low half)
      g1[3] = (int)(256u << 16);          // tile_dim0 = 256   (bits 127:112)
      g1[4] = 16;                         // tile_dim1 = 16; tile_dim2 = 0
      g1[5] = TT * NN;                    // tensor_dim0_stride = 32768
      g1[6] = 0;
      g1[7] = 0;
      i32x4 g2 = {}, g3 = {};
#if __clang_major__ >= 23
      i32x8 g4 = {};
      __builtin_amdgcn_tensor_load_to_lds(g0, g1, g2, g3, g4, 0);
#else
      __builtin_amdgcn_tensor_load_to_lds(g0, g1, g2, g3, 0);
#endif
    }
#endif

    // ---- (1) hs = [h;c] @ We + be : 16 x 128, K = 256. waves 0..7 -> col tiles
    if (wave < 8) {
      int s0 = wave * 16;
      v8f acc = {};
#pragma unroll
      for (int kc = 0; kc < 8; ++kc) {
        v16bf a  = load_frag(&hc_b[0][0], 2 * MM, kc * 32, lane);
        v16bf bb = load_frag(WeT + (size_t)s0 * (2 * MM), 2 * MM, kc * 32, lane);
        acc = wmma_bf16(a, bb, acc);
      }
      int s = s0 + (lane & 15);
      float bes = be[s];
#pragma unroll
      for (int v = 0; v < 8; ++v) {
        int m = v + ((lane >> 4) << 3);
        scratch[m][HS_OFF + s] = acc[v] + bes;
      }
    }
    __syncthreads();

    // ---- (2) e[bi][n] = sum_s tanh(hs[bi][s] + ux[b,n,s]) * ve[s] + bv
    //      512 threads: n = tid&255, each handles 8 rows
    {
      int n = tid & 255;
      int bi0 = (tid >> 8) * 8;
      for (int r = 0; r < 8; ++r) {
        int bi = bi0 + r;
        const bf16_t* uxp = ux + ((size_t)(b0 + bi) * NN + n) * TT;
        if (r + 1 < 8)
          __builtin_prefetch(ux + ((size_t)(b0 + bi + 1) * NN + n) * TT, 0, 1);
        float acc = bvv;
#pragma unroll 4
        for (int sc = 0; sc < TT; sc += 8) {
          v8bf uv = *(const v8bf*)(uxp + sc);
#pragma unroll
          for (int k = 0; k < 8; ++k)
            acc += ftanh(scratch[bi][HS_OFF + sc + k] + (float)uv[k]) * ve_s[sc + k];
        }
        scratch[bi][n] = acc;  // e region: cols [0..255]
      }
    }
#if HAVE_TDM
    if (wave == 15) wait_tensorcnt0();   // x_t tile resident before barrier
#endif
    __syncthreads();

    // ---- (3) softmax over N + x_tilde (bf16). wave w -> row w
    {
      int bi = wave;
      float ev[8], mx = -1e30f;
#pragma unroll
      for (int j = 0; j < 8; ++j) { ev[j] = scratch[bi][lane + 32 * j]; mx = fmaxf(mx, ev[j]); }
#pragma unroll
      for (int off = 16; off >= 1; off >>= 1) mx = fmaxf(mx, __shfl_xor(mx, off, 32));
      float al[8], sum = 0.0f;
#pragma unroll
      for (int j = 0; j < 8; ++j) { al[j] = fexp2(1.44269504f * (ev[j] - mx)); sum += al[j]; }
#pragma unroll
      for (int off = 16; off >= 1; off >>= 1) sum += __shfl_xor(sum, off, 32);
      float rs = frcp(sum);
#if HAVE_TDM
#pragma unroll
      for (int j = 0; j < 8; ++j) {
        int n = lane + 32 * j;
        xt_b[bi][n] = (bf16_t)(al[j] * rs * Xs_f[bi][n]);
      }
#else
      const float* xrow = X + ((size_t)(b0 + bi) * TT + t) * NN;
#pragma unroll
      for (int j = 0; j < 8; ++j) {
        int n = lane + 32 * j;
        xt_b[bi][n] = (bf16_t)(al[j] * rs * xrow[n]);
      }
#endif
    }
    __syncthreads();

    // ---- (4) gates = x_tilde @ Wx + h @ Wh + b : 16 x 512. wave w -> 2 col tiles
    {
      int j0 = wave * 32;
      v8f acc[2] = {{}, {}};
#pragma unroll
      for (int kc = 0; kc < 8; ++kc) {                  // K over x_tilde (256)
        v16bf a = load_frag(&xt_b[0][0], NN, kc * 32, lane);
#pragma unroll
        for (int q = 0; q < 2; ++q) {
          v16bf bb = load_frag(WxT + (size_t)(j0 + q * 16) * NN, NN, kc * 32, lane);
          acc[q] = wmma_bf16(a, bb, acc[q]);
        }
      }
#pragma unroll
      for (int kc = 0; kc < 4; ++kc) {                  // K over h (128)
        v16bf a = load_frag(&hc_b[0][0], 2 * MM, kc * 32, lane);
#pragma unroll
        for (int q = 0; q < 2; ++q) {
          v16bf bb = load_frag(WhT + (size_t)(j0 + q * 16) * MM, MM, kc * 32, lane);
          acc[q] = wmma_bf16(a, bb, acc[q]);
        }
      }
#pragma unroll
      for (int q = 0; q < 2; ++q) {
        int j = j0 + q * 16 + (lane & 15);
        float bj = gb[j];
#pragma unroll
        for (int v = 0; v < 8; ++v) {
          int m = v + ((lane >> 4) << 3);
          scratch[m][j] = acc[q][v] + bj;               // gates overwrite scratch
        }
      }
    }
    __syncthreads();

    // ---- (5) LSTM pointwise + state update + output
    {
#pragma unroll
      for (int k = 0; k < 4; ++k) {
        int idx = tid + 512 * k;          // 0..2047
        int bi = idx >> 7;
        int j  = idx & 127;
        float ig = scratch[bi][j];
        float fg = scratch[bi][MM + j];
        float gg = scratch[bi][2 * MM + j];
        float og = scratch[bi][3 * MM + j];
        float cn = fsig(fg) * c_s[bi][j] + fsig(ig) * ftanh(gg);
        float hn = fsig(og) * ftanh(cn);
        c_s[bi][j] = cn;
        hc_b[bi][j]      = (bf16_t)hn;
        hc_b[bi][MM + j] = (bf16_t)cn;
        out[((size_t)(b0 + bi) * TT + t) * MM + j] = hn;
      }
    }
    __syncthreads();
  }
}

// ---------------- launcher ----------------
extern "C" void kernel_launch(void* const* d_in, const int* in_sizes, int n_in,
                              void* d_out, int out_size, void* d_ws, size_t ws_size,
                              hipStream_t stream) {
  const float* X  = (const float*)d_in[0];
  const float* We = (const float*)d_in[1];
  const float* be = (const float*)d_in[2];
  const float* Ue = (const float*)d_in[3];
  const float* bu = (const float*)d_in[4];
  const float* ve = (const float*)d_in[5];
  const float* bv = (const float*)d_in[6];
  const float* Wx = (const float*)d_in[7];
  const float* Wh = (const float*)d_in[8];
  const float* gb = (const float*)d_in[9];
  float* out = (float*)d_out;

  char* ws = (char*)d_ws;
  size_t off = 0;
  bf16_t* ux  = (bf16_t*)(ws + off); off += (size_t)BB * NN * TT * 2;  // 64 MB
  bf16_t* UeT = (bf16_t*)(ws + off); off += 128 * 128 * 2;
  bf16_t* WeT = (bf16_t*)(ws + off); off += 128 * 256 * 2;
  bf16_t* WxT = (bf16_t*)(ws + off); off += 512 * 256 * 2;
  bf16_t* WhT = (bf16_t*)(ws + off); off += 512 * 128 * 2;

  prep_weights<<<192, 256, 0, stream>>>(We, Ue, Wx, Wh, UeT, WeT, WxT, WhT);
  ux_gemm<<<(BB * NN) / 16, 256, 0, stream>>>(X, bu, UeT, ux);
  darnn_scan<<<BB / 16, 512, 0, stream>>>(X, be, ve, bv, gb, ux, WeT, WxT, WhT, out);
}
